// MultiHeadAttention_79826262163594
// MI455X (gfx1250) — matrix operand order repaired and checked by reading, not compiled, not run on hardware
//
#include <hip/hip_runtime.h>

typedef __bf16 v16bf __attribute__((ext_vector_type(16)));
typedef __bf16 v8bf  __attribute__((ext_vector_type(8)));
typedef __bf16 v4bf  __attribute__((ext_vector_type(4)));
typedef float  v8f   __attribute__((ext_vector_type(8)));
typedef unsigned int u32x4 __attribute__((ext_vector_type(4)));
typedef int i32x4 __attribute__((ext_vector_type(4)));
typedef int i32x8 __attribute__((ext_vector_type(8)));

#define WMMA_BF16(a, b, c) \
  __builtin_amdgcn_wmma_f32_16x16x32_bf16(false, (a), false, (b), (short)0, (c), false, false)

static __device__ __forceinline__ v16bf cat8(v8bf lo, v8bf hi8) {
  return __builtin_shufflevector(lo, hi8, 0, 1, 2, 3, 4, 5, 6, 7, 8, 9, 10, 11, 12, 13, 14, 15);
}

// ---------------------------------------------------------------------------
// TDM: async DMA of a 128x32 bf16 tile (row stride 1024 elems) into LDS with
// 4-DWORD padding after every 16 DWORDs -> padded LDS rows of 40 bf16 (80 B,
// 16B-aligned for ds_load_b128 fragment loads).
// ---------------------------------------------------------------------------
static __device__ __forceinline__ void tdm_load_tile_128x32(unsigned lds_byte,
                                                            const __bf16* g) {
  const unsigned long long ga = (unsigned long long)(uintptr_t)g;
  u32x4 g0;
  g0.x = 1u;                                                  // count=1 (user D#)
  g0.y = lds_byte;                                            // lds_addr (bytes)
  g0.z = (unsigned)ga;                                        // global_addr[31:0]
  g0.w = (unsigned)((ga >> 32) & 0x01ffffffu) | (2u << 30);   // addr[56:32] | type=2
  i32x8 g1;
  // data_size=1 (2B), pad_enable=1, pad_interval=3 (16 DW), pad_amount=3 (4 DW)
  g1[0] = (1 << 16) | (1 << 20) | (3 << 22) | (3 << 25);
  g1[1] = (int)((1024u & 0xffffu) << 16);                     // tensor_dim0 = 1024
  g1[2] = (int)((1024u >> 16) | ((4096u & 0xffffu) << 16));   // dim0 hi | tensor_dim1 lo
  g1[3] = (int)((4096u >> 16) | (32u << 16));                 // dim1 hi | tile_dim0 = 32
  g1[4] = 128;                                                // tile_dim1=128, tile_dim2=0
  g1[5] = 1024;                                               // tensor_dim0_stride (elems)
  g1[6] = 0;
  g1[7] = 0;
  const i32x4 z4 = {0, 0, 0, 0};
#if __clang_major__ >= 23
  const i32x8 z8 = {0, 0, 0, 0, 0, 0, 0, 0};
  __builtin_amdgcn_tensor_load_to_lds(g0, g1, z4, z4, z8, 0);
#else
  __builtin_amdgcn_tensor_load_to_lds(g0, g1, z4, z4, 0);
#endif
}

// ---------------------------------------------------------------------------
// f32 -> bf16 streaming convert (4 elems/thread).
// ---------------------------------------------------------------------------
__global__ __launch_bounds__(256) void conv_kernel(const float* __restrict__ in,
                                                   __bf16* __restrict__ out) {
  const size_t i = ((size_t)blockIdx.x * 256 + threadIdx.x) * 4;
  const float4 v = *(const float4*)(in + i);
  v4bf o;
  o[0] = (__bf16)v.x; o[1] = (__bf16)v.y; o[2] = (__bf16)v.z; o[3] = (__bf16)v.w;
  *(v4bf*)(out + i) = o;
}

// ---------------------------------------------------------------------------
// f32 [1024,1024] -> bf16 transposed [1024,1024], tiled via LDS.
// ---------------------------------------------------------------------------
__global__ __launch_bounds__(256) void transp_conv_kernel(const float* __restrict__ W,
                                                          __bf16* __restrict__ Wt) {
  __shared__ __bf16 t[32][33];
  const int bx = blockIdx.x * 32, by = blockIdx.y * 32;
  const int x = threadIdx.x, y = threadIdx.y;  // block (32, 8)
#pragma unroll
  for (int yy = y; yy < 32; yy += 8)
    t[yy][x] = (__bf16)W[(size_t)(by + yy) * 1024 + bx + x];
  __syncthreads();
#pragma unroll
  for (int yy = y; yy < 32; yy += 8)
    Wt[(size_t)(bx + yy) * 1024 + by + x] = t[x][yy];
}

// ---------------------------------------------------------------------------
// GEMM: C[4096,1024] = A(bf16)[4096,1024] @ Wt(bf16,[n][k])^T + bias, *scale
// Tiles 128x128, K-step 32, TDM double-buffered staging, 8 waves (2x4).
// MODE 0: bf16 out scattered to [B,H,S,HD]  (Q, K)
// MODE 1: bf16 out scattered to [B,H,HD,S]  (V, pre-transposed)
// MODE 2: f32 out row-major [M,N]           (final projection)
// ---------------------------------------------------------------------------
template <int MODE>
__global__ __launch_bounds__(256) void gemm_kernel(
    const __bf16* __restrict__ A, const __bf16* __restrict__ Bt,
    const float* __restrict__ bias, void* __restrict__ outp, float scale) {
  constexpr int LD = 40;  // 32 data + 8 pad bf16 per row (80 B)
  __shared__ __align__(16) __bf16 As[2][128 * LD];
  __shared__ __align__(16) __bf16 Bs[2][128 * LD];

  const int tid = threadIdx.x;
  const int wid = tid >> 5, lane = tid & 31;
  const int r = lane & 15, hi = lane >> 4;
  const int wave_m = wid >> 2;  // 0..1 (64 rows)
  const int wave_n = wid & 3;   // 0..3 (32 cols)
  const int m0 = blockIdx.y * 128;
  const int n0 = blockIdx.x * 128;

  const __bf16* Abase = A + (size_t)m0 * 1024;
  const __bf16* Bbase = Bt + (size_t)n0 * 1024;

  v8f acc[4][2];
#pragma unroll
  for (int i = 0; i < 4; ++i)
#pragma unroll
    for (int j = 0; j < 2; ++j) acc[i][j] = {};

  // Prologue: DMA tile 0 into buffer 0.
  if (wid == 0) {
    tdm_load_tile_128x32((unsigned)(uintptr_t)&As[0][0], Abase);
    tdm_load_tile_128x32((unsigned)(uintptr_t)&Bs[0][0], Bbase);
    __builtin_amdgcn_s_wait_tensorcnt(0);
  }
  __syncthreads();

  for (int k0 = 0; k0 < 1024; k0 += 32) {
    const int cur = (k0 >> 5) & 1;
    const int nxt = cur ^ 1;
    if (k0 + 32 < 1024 && wid == 0) {  // kick off next tile DMA (overlaps WMMA)
      tdm_load_tile_128x32((unsigned)(uintptr_t)&As[nxt][0], Abase + k0 + 32);
      tdm_load_tile_128x32((unsigned)(uintptr_t)&Bs[nxt][0], Bbase + k0 + 32);
    }

    v16bf af[4], bfr[2];
#pragma unroll
    for (int i = 0; i < 4; ++i) {
      // A frag (16x32): lane row = m; K: {hi*8+0..7, 16+hi*8+0..7}
      const __bf16* p = &As[cur][(wave_m * 64 + i * 16 + r) * LD];
      af[i] = cat8(*(const v8bf*)(p + hi * 8), *(const v8bf*)(p + 16 + hi * 8));
    }
#pragma unroll
    for (int j = 0; j < 2; ++j) {
      // B frag (32x16): lane col = n; K: {hi*8+0..7, 16+hi*8+0..7}
      const __bf16* p = &Bs[cur][(wave_n * 32 + j * 16 + r) * LD] + hi * 8;
      bfr[j] = cat8(*(const v8bf*)p, *(const v8bf*)(p + 16));
    }
#pragma unroll
    for (int i = 0; i < 4; ++i)
#pragma unroll
      for (int j = 0; j < 2; ++j) acc[i][j] = WMMA_BF16(af[i], bfr[j], acc[i][j]);

    if (k0 + 32 < 1024 && wid == 0) __builtin_amdgcn_s_wait_tensorcnt(0);
    __syncthreads();
  }

  // Epilogue. C/D layout: VGPR v -> row (v + 8*hi), lane r -> col.
#pragma unroll
  for (int i = 0; i < 4; ++i) {
#pragma unroll
    for (int j = 0; j < 2; ++j) {
      const int mr = m0 + wave_m * 64 + i * 16;
      const int nc = n0 + wave_n * 32 + j * 16 + r;
      const float bv = bias[nc];
#pragma unroll
      for (int v = 0; v < 8; ++v) {
        const int m = mr + v + hi * 8;
        const float val = (acc[i][j][v] + bv) * scale;
        if (MODE == 2) {
          ((float*)outp)[(size_t)m * 1024 + nc] = val;
        } else {
          const int b = m >> 11, s = m & 2047;  // m = b*S + s
          const int h = nc >> 6, hd = nc & 63;  // n = h*HD + hd
          size_t idx;
          if (MODE == 0)
            idx = (((size_t)b * 16 + h) * 2048 + s) * 64 + hd;  // [B,H,S,HD]
          else
            idx = (((size_t)b * 16 + h) * 64 + hd) * 2048 + s;  // [B,H,HD,S]
          ((__bf16*)outp)[idx] = (__bf16)val;
        }
      }
    }
  }
}

// ---------------------------------------------------------------------------
// Flash attention, transposed: S^T = K @ Q^T, O^T = V^T @ P^T.
// One wave = 16 queries (lane r = one query): softmax stats are per-lane
// scalars, only xor-16 shuffles cross lanes. No LDS, no barriers.
// Qh: [B,H,S,HD] bf16 (pre-scaled 1/8), Kh: [B,H,S,HD], Vt: [B,H,HD,S],
// AOb: [B,S,D] bf16 (feeds final GEMM).
// ---------------------------------------------------------------------------
__global__ __launch_bounds__(256) void attn_kernel(
    const __bf16* __restrict__ Qh, const __bf16* __restrict__ Kh,
    const __bf16* __restrict__ Vt, __bf16* __restrict__ AOb,
    const int* __restrict__ maskp) {
  const int tid = threadIdx.x;
  const int wid = tid >> 5, lane = tid & 31;
  const int r = lane & 15, hi = lane >> 4;
  const int q0 = (blockIdx.x * 8 + wid) * 16;
  const int bh = blockIdx.y;

  const __bf16* Qb = Qh + (size_t)bh * 2048 * 64;
  const __bf16* Kb = Kh + (size_t)bh * 2048 * 64;
  const __bf16* Vb = Vt + (size_t)bh * 64 * 2048;
  const bool causal = (*maskp != 0);
  const int kmax = causal ? (q0 + 16) : 2048;

  // Q as B-fragments of S^T: lane col = query q0+r; hd = f*32 + {hi*8, 16+hi*8}+e.
  v16bf qf[2];
#pragma unroll
  for (int f = 0; f < 2; ++f) {
    const __bf16* p = Qb + (size_t)(q0 + r) * 64 + f * 32 + hi * 8;
    qf[f] = cat8(*(const v8bf*)p, *(const v8bf*)(p + 16));
  }

  float m_run = -1e30f, l_run = 0.f;
  v8f Oacc[4];
#pragma unroll
  for (int d = 0; d < 4; ++d) Oacc[d] = {};

  for (int k0 = 0; k0 < kmax; k0 += 64) {
    // ---- S^T tiles: rows = keys (v+8*hi within tile t), col = query r.
    v8f sc[4];
#pragma unroll
    for (int t = 0; t < 4; ++t) {
      v8f s = {};
#pragma unroll
      for (int f = 0; f < 2; ++f) {
        // K as A-frag: lane row = key k0+16t+r; K-dim(hd) = f*32 + {hi*8, 16+hi*8}+e
        const __bf16* p = Kb + (size_t)(k0 + 16 * t + r) * 64 + f * 32;
        const v16bf kf = cat8(*(const v8bf*)(p + hi * 8), *(const v8bf*)(p + 16 + hi * 8));
        s = WMMA_BF16(kf, qf[f], s);
      }
      sc[t] = s;
    }
    if (causal) {
      const int q = q0 + r;
#pragma unroll
      for (int t = 0; t < 4; ++t)
#pragma unroll
        for (int v = 0; v < 8; ++v)
          if (k0 + 16 * t + v + 8 * hi > q) sc[t][v] = -1e9f;
    }
    // ---- online softmax: per-lane (per-query) scalars; one xor-16 combine.
    float tmax = -1e30f;
#pragma unroll
    for (int t = 0; t < 4; ++t)
#pragma unroll
      for (int v = 0; v < 8; ++v) tmax = fmaxf(tmax, sc[t][v]);
    tmax = fmaxf(tmax, __shfl_xor(tmax, 16, 32));
    const float mnew = fmaxf(m_run, tmax);
    const float alpha = __expf(m_run - mnew);
    float lsum = 0.f;
#pragma unroll
    for (int t = 0; t < 4; ++t)
#pragma unroll
      for (int v = 0; v < 8; ++v) {
        const float e = __expf(sc[t][v] - mnew);
        sc[t][v] = e;
        lsum += e;
      }
    lsum += __shfl_xor(lsum, 16, 32);
    l_run = l_run * alpha + lsum;
    m_run = mnew;
#pragma unroll
    for (int d = 0; d < 4; ++d)
#pragma unroll
      for (int v = 0; v < 8; ++v) Oacc[d][v] *= alpha;

    // ---- P^T B-frags straight from S^T D-layout, no shuffle.
    // Element e of frag f holds key 32f+8*hi+e (e<8, tile 2f) and
    // 32f+16+8*hi+(e-8) (e>=8, tile 2f+1).
    v16bf pf[2];
#pragma unroll
    for (int f = 0; f < 2; ++f) {
      union { v8bf v; int i[4]; } own, oth;
#pragma unroll
      for (int e = 0; e < 8; ++e) {
        own.v[e] = (__bf16)sc[2 * f][e];
        oth.v[e] = (__bf16)sc[2 * f + 1][e];
      }
      pf[f] = cat8(own.v, oth.v);
    }
    // ---- O^T += V^T @ P^T  (V^T A-frags contiguous from [B,H,HD,S]).
#pragma unroll
    for (int d = 0; d < 4; ++d)
#pragma unroll
      for (int f = 0; f < 2; ++f) {
        const __bf16* vp = Vb + (size_t)(16 * d + r) * 2048 + k0 + f * 32;
        const v16bf vf = cat8(*(const v8bf*)(vp + hi * 8), *(const v8bf*)(vp + 16 + hi * 8));
        Oacc[d] = WMMA_BF16(vf, pf[f], Oacc[d]);
      }
  }

  // ---- epilogue: lane r owns query q0+r -> 4 contiguous v8bf stores.
  const int b = bh >> 4, h = bh & 15;
  const float inv_l = 1.f / l_run;
  __bf16* orow = AOb + ((size_t)b * 2048 + q0 + r) * 1024 + h * 64;
#pragma unroll
  for (int d = 0; d < 4; ++d) {
    v8bf o;
#pragma unroll
    for (int v = 0; v < 8; ++v) o[v] = (__bf16)(Oacc[d][v] * inv_l);
    *(v8bf*)(orow + 16 * d + 8 * hi) = o;
  }
}

// ---------------------------------------------------------------------------
extern "C" void kernel_launch(void* const* d_in, const int* in_sizes, int n_in,
                              void* d_out, int out_size, void* d_ws, size_t ws_size,
                              hipStream_t stream) {
  (void)in_sizes; (void)n_in; (void)out_size; (void)ws_size;
  const float* X  = (const float*)d_in[0];
  const float* Wq = (const float*)d_in[1];
  const float* bq = (const float*)d_in[2];
  const float* Wk = (const float*)d_in[3];
  const float* bk = (const float*)d_in[4];
  const float* Wv = (const float*)d_in[5];
  const float* bv = (const float*)d_in[6];
  const float* Wo = (const float*)d_in[7];
  const float* bo = (const float*)d_in[8];
  const int*  msk = (const int*)d_in[9];

  char* ws = (char*)d_ws;
  const size_t MB = 1024 * 1024;
  __bf16* Xb  = (__bf16*)(ws);             // 8 MB  [4096,1024]
  __bf16* Wqt = (__bf16*)(ws + 8 * MB);    // 2 MB  [n][k]
  __bf16* Wkt = (__bf16*)(ws + 10 * MB);
  __bf16* Wvt = (__bf16*)(ws + 12 * MB);
  __bf16* Wot = (__bf16*)(ws + 14 * MB);
  __bf16* Qh  = (__bf16*)(ws + 16 * MB);   // 8 MB [B,H,S,HD]
  __bf16* Kh  = (__bf16*)(ws + 24 * MB);   // 8 MB [B,H,S,HD]
  __bf16* Vt  = (__bf16*)(ws + 32 * MB);   // 8 MB [B,H,HD,S]
  __bf16* AOb = (__bf16*)(ws + 40 * MB);   // 8 MB [B,S,D]

  conv_kernel<<<4096, 256, 0, stream>>>(X, Xb);
  const dim3 tg(32, 32), tb(32, 8);
  transp_conv_kernel<<<tg, tb, 0, stream>>>(Wq, Wqt);
  transp_conv_kernel<<<tg, tb, 0, stream>>>(Wk, Wkt);
  transp_conv_kernel<<<tg, tb, 0, stream>>>(Wv, Wvt);
  transp_conv_kernel<<<tg, tb, 0, stream>>>(Wo, Wot);

  const dim3 gg(8, 32), gb(256);
  gemm_kernel<0><<<gg, gb, 0, stream>>>(Xb, Wqt, bq, (void*)Qh, 0.125f);  // 1/sqrt(64)
  gemm_kernel<0><<<gg, gb, 0, stream>>>(Xb, Wkt, bk, (void*)Kh, 1.0f);
  gemm_kernel<1><<<gg, gb, 0, stream>>>(Xb, Wvt, bv, (void*)Vt, 1.0f);
  attn_kernel<<<dim3(16, 32), 256, 0, stream>>>(Qh, Kh, Vt, AOb, msk);
  gemm_kernel<2><<<gg, gb, 0, stream>>>(AOb, Wot, bo, d_out, 1.0f);
}
